// Transformer_69681549410849
// MI455X (gfx1250) — matrix operand order repaired and checked by reading, not compiled, not run on hardware
//
#include <hip/hip_runtime.h>
#include <hip/hip_bf16.h>

typedef __attribute__((ext_vector_type(16))) __bf16 v16bf;
typedef __attribute__((ext_vector_type(8)))  float  v8f;

#define B_  2
#define S_  1024
#define D_  4096
#define H_  32
#define HD_ 128
#define AL_ 10
#define SK_ (S_ + AL_)          // 1034 k/v rows per batch (adapter + sequence)

// ---------------------------------------------------------------------------
// gfx1250 async global->LDS path (ASYNCcnt), guarded so the file compiles on
// toolchains that do not expose the builtin (falls back to ds stores).
// Prototype (from hipcc diagnostic): param0 = int4 addrspace(1)*, then the
// LDS destination, immediate offset, cpol.
// ---------------------------------------------------------------------------
#if defined(__has_builtin)
#if __has_builtin(__builtin_amdgcn_global_load_async_to_lds_b128)
#define USE_ASYNC_LDS 1
#endif
#endif
#ifndef USE_ASYNC_LDS
#define USE_ASYNC_LDS 0
#endif

typedef int v4i_vs __attribute__((vector_size(4 * sizeof(int))));
typedef __attribute__((address_space(1))) v4i_vs g1_v4i;   // global int4
typedef __attribute__((address_space(3))) v4i_vs l3_v4i;   // LDS int4

__device__ inline void wait_async_lds() {
#if USE_ASYNC_LDS
#if __has_builtin(__builtin_amdgcn_s_wait_asynccnt)
  __builtin_amdgcn_s_wait_asynccnt(0);
#else
  asm volatile("s_wait_asynccnt 0x0" ::: "memory");
#endif
#endif
}

// ---------------------------------------------------------------------------
// WMMA helpers
// ---------------------------------------------------------------------------
__device__ inline v8f wmma_bf16(v16bf a, v16bf b, v8f c) {
  return __builtin_amdgcn_wmma_f32_16x16x32_bf16(false, a, false, b,
                                                 (short)0, c, false, false);
}

// A fragment: 16x32 bf16 tile, row-major in LDS with row stride `ld` elems.
// Lane (lr = lane&15) holds row M=lr; half = lane>>4 selects K groups.
// Element e of v16bf maps to K = (e/8)*16 + half*8 + (e%8)  -> two b128 loads.
__device__ inline v16bf ld_frag_a(const __bf16* p, int ld, int lane) {
  const int lr = lane & 15, half = (lane >> 4) & 1;
  const __bf16* q = p + lr * ld + half * 8;
  v16bf f;
  ((uint4*)&f)[0] = *(const uint4*)(q);
  ((uint4*)&f)[1] = *(const uint4*)(q + 16);
  return f;
}

// B fragment: Bt stored [N][K] row-major, 16(N)x32(K) tile.
// Lane holds column N=lr; element e maps to K = (e/8)*16 + half*8 + (e%8).
__device__ inline v16bf ld_frag_b(const __bf16* p, int ld, int lane) {
  const int lr = lane & 15, half = (lane >> 4) & 1;
  const __bf16* q = p + lr * ld + half * 8;
  v16bf f;
  ((uint4*)&f)[0] = *(const uint4*)(q);
  ((uint4*)&f)[1] = *(const uint4*)(q + 16);
  return f;
}

__device__ inline float rmax16(float x) {
#pragma unroll
  for (int o = 1; o < 16; o <<= 1) x = fmaxf(x, __shfl_xor(x, o, 32));
  return x;
}
__device__ inline float rsum16(float x) {
#pragma unroll
  for (int o = 1; o < 16; o <<= 1) x += __shfl_xor(x, o, 32);
  return x;
}

// ---------------------------------------------------------------------------
// fp32 -> bf16 convert
// ---------------------------------------------------------------------------
__global__ __launch_bounds__(256) void convert_f32_bf16(
    const float* __restrict__ in, __bf16* __restrict__ out, int n) {
  int i = blockIdx.x * 256 + threadIdx.x;
  if (i < n) out[i] = (__bf16)in[i];
}

// fp32 [K][N] -> bf16 [N][K] (weight transpose so GEMM reads K-contiguous)
__global__ __launch_bounds__(256) void transpose_f32_bf16(
    const float* __restrict__ in, __bf16* __restrict__ out, int K, int N) {
  __shared__ float tile[32][33];
  const int kb = blockIdx.y * 32, nb = blockIdx.x * 32;
  const int tx = threadIdx.x & 31, ty = threadIdx.x >> 5;  // 32 x 8
#pragma unroll
  for (int i = 0; i < 32; i += 8)
    tile[ty + i][tx] = in[(size_t)(kb + ty + i) * N + nb + tx];
  __syncthreads();
#pragma unroll
  for (int i = 0; i < 32; i += 8)
    out[(size_t)(nb + ty + i) * K + kb + tx] = (__bf16)tile[tx][ty + i];
}

// ---------------------------------------------------------------------------
// Tiled WMMA GEMM: C[M][N] = A[M][K] (bf16, row-major) x Bt[N][K] (bf16).
// 256 threads = 8 waves in 4x2 grid; each wave computes 32x64 (2x4 tiles).
// Double-buffered LDS staging: the next K-tile is fetched (async-to-LDS where
// available) while WMMAs consume the current one.
// Output rows remapped (b, s) -> b*dstRowsPerBatch + dstRowOff + s so K/V
// land after the adapter rows.
// ---------------------------------------------------------------------------
template <bool F32OUT>
__global__ __launch_bounds__(256) void gemm_wmma(
    const __bf16* __restrict__ A, const __bf16* __restrict__ Bt,
    void* __restrict__ Cout, int M, int N, int K,
    int srcRowsPerBatch, int dstRowsPerBatch, int dstRowOff) {
  __shared__ __align__(16) __bf16 As[2][128][64];
  __shared__ __align__(16) __bf16 Bs[2][128][64];
  const int tid = threadIdx.x;
  const int lane = tid & 31;
  const int w = tid >> 5;
  const int lr = lane & 15, half = (lane >> 4) & 1;
  const int mBase = blockIdx.y * 128;
  const int nBase = blockIdx.x * 128;
  const int mOff = (w >> 1) * 32;
  const int nOff = (w & 1) * 64;

  v8f acc[2][4];
#pragma unroll
  for (int i = 0; i < 2; i++)
#pragma unroll
    for (int j = 0; j < 4; j++) acc[i][j] = (v8f){};

  const int sr = tid >> 1, sh = tid & 1;  // staging: row, 32-elem half
  const __bf16* gA = A + (size_t)(mBase + sr) * K + sh * 32;
  const __bf16* gB = Bt + (size_t)(nBase + sr) * K + sh * 32;

  auto stage_tile = [&](int buf, int k0) {
#if USE_ASYNC_LDS
#pragma unroll
    for (int j = 0; j < 4; ++j) {
      __builtin_amdgcn_global_load_async_to_lds_b128(
          (g1_v4i*)(gA + k0) + j, (l3_v4i*)(&As[buf][sr][sh * 32]) + j, 0, 0);
      __builtin_amdgcn_global_load_async_to_lds_b128(
          (g1_v4i*)(gB + k0) + j, (l3_v4i*)(&Bs[buf][sr][sh * 32]) + j, 0, 0);
    }
#else
    const uint4* ga = (const uint4*)(gA + k0);
    uint4* la = (uint4*)(&As[buf][sr][sh * 32]);
    la[0] = ga[0]; la[1] = ga[1]; la[2] = ga[2]; la[3] = ga[3];
    const uint4* gb = (const uint4*)(gB + k0);
    uint4* lb = (uint4*)(&Bs[buf][sr][sh * 32]);
    lb[0] = gb[0]; lb[1] = gb[1]; lb[2] = gb[2]; lb[3] = gb[3];
#endif
  };

  stage_tile(0, 0);
  wait_async_lds();
  __syncthreads();

  int buf = 0;
  for (int k0 = 0; k0 < K; k0 += 64) {
    if (k0 + 64 < K) stage_tile(buf ^ 1, k0 + 64);  // overlap with compute
#pragma unroll
    for (int kk = 0; kk < 2; ++kk) {
      v16bf af[2], bf[4];
#pragma unroll
      for (int tm = 0; tm < 2; tm++)
        af[tm] = ld_frag_a(&As[buf][mOff + tm * 16][kk * 32], 64, lane);
#pragma unroll
      for (int tn = 0; tn < 4; tn++)
        bf[tn] = ld_frag_b(&Bs[buf][nOff + tn * 16][kk * 32], 64, lane);
#pragma unroll
      for (int tm = 0; tm < 2; tm++)
#pragma unroll
        for (int tn = 0; tn < 4; tn++)
          acc[tm][tn] = wmma_bf16(af[tm], bf[tn], acc[tm][tn]);
    }
    wait_async_lds();   // drain this iteration's async fills
    __syncthreads();    // all waves done reading buf / filling buf^1
    buf ^= 1;
  }

  // epilogue: C layout -> row = r + 8*half, col = lr within each 16x16 tile
#pragma unroll
  for (int tm = 0; tm < 2; tm++)
#pragma unroll
    for (int tn = 0; tn < 4; tn++)
#pragma unroll
      for (int r = 0; r < 8; r++) {
        const int row = mBase + mOff + tm * 16 + r + 8 * half;
        const int col = nBase + nOff + tn * 16 + lr;
        const int b = row / srcRowsPerBatch, s = row % srcRowsPerBatch;
        const size_t drow = (size_t)b * dstRowsPerBatch + dstRowOff + s;
        if (F32OUT)
          ((float*)Cout)[drow * N + col] = acc[tm][tn][r];
        else
          ((__bf16*)Cout)[drow * N + col] = (__bf16)acc[tm][tn][r];
      }
}

// ---------------------------------------------------------------------------
// adapter @ wk / wv  (tiny: 10 x 4096 outputs), broadcast into both batches
// ---------------------------------------------------------------------------
__global__ __launch_bounds__(256) void adapter_kv(
    const float* __restrict__ adapter, const __bf16* __restrict__ wkT,
    const __bf16* __restrict__ wvT, __bf16* __restrict__ kb,
    __bf16* __restrict__ vb) {
  const int i = blockIdx.x * 256 + threadIdx.x;
  if (i >= AL_ * D_) return;
  const int al = i / D_, c = i % D_;
  const float* arow = adapter + (size_t)al * D_;
  const __bf16* kr = wkT + (size_t)c * D_;
  const __bf16* vr = wvT + (size_t)c * D_;
  float a = 0.f, v = 0.f;
  for (int k = 0; k < D_; k++) {
    const float xv = arow[k];
    a += xv * (float)kr[k];
    v += xv * (float)vr[k];
  }
#pragma unroll
  for (int bb = 0; bb < B_; bb++) {
    kb[((size_t)bb * SK_ + al) * D_ + c] = (__bf16)a;
    vb[((size_t)bb * SK_ + al) * D_ + c] = (__bf16)v;
  }
}

// ---------------------------------------------------------------------------
// RoPE on bf16 [rows][D]: pairs (2i, 2i+1) within each head
// ---------------------------------------------------------------------------
__global__ __launch_bounds__(256) void rope_kernel(
    __bf16* __restrict__ buf, const float* __restrict__ cosb,
    const float* __restrict__ sinb, int dstRowsPerBatch, int dstRowOff) {
  const size_t i = (size_t)blockIdx.x * 256 + threadIdx.x;  // pair id
  const int pairsPerRow = D_ / 2;
  const int r = (int)(i / pairsPerRow);  // 0 .. B*S-1
  const int p = (int)(i % pairsPerRow);
  const int bb = r / S_, s = r % S_;
  const int hpi = p % (HD_ / 2);
  const int c0 = (p / (HD_ / 2)) * HD_ + 2 * hpi;
  const size_t row = (size_t)bb * dstRowsPerBatch + dstRowOff + s;
  const float c = cosb[s * (HD_ / 2) + hpi];
  const float sn = sinb[s * (HD_ / 2) + hpi];
  __bf16* pr = buf + row * D_ + c0;
  const float v0 = (float)pr[0], v1 = (float)pr[1];
  pr[0] = (__bf16)(v0 * c - v1 * sn);
  pr[1] = (__bf16)(v0 * sn + v1 * c);
}

// ---------------------------------------------------------------------------
// Flash attention with adapter-prefix split softmax.
// Block = (q-tile of 128, head, batch); 8 waves, wave w owns q-rows [16w,16w+16).
// ---------------------------------------------------------------------------
__global__ __launch_bounds__(256) void attn_kernel(
    const __bf16* __restrict__ qb,  // [B][S][D]
    const __bf16* __restrict__ kb,  // [B][SK][D]
    const __bf16* __restrict__ vb,  // [B][SK][D]
    __bf16* __restrict__ ob,        // [B][S][D]
    const float* __restrict__ gate1, const float* __restrict__ gate2,
    const int* __restrict__ vsp) {
  __shared__ __align__(16) __bf16 Qs[128][128];
  __shared__ __align__(16) __bf16 Ks[32][128];
  __shared__ __align__(16) __bf16 Vt[128][40];  // padded stride: 80B, 16B-aligned
  __shared__ __align__(16) __bf16 Ps[128][32];
  __shared__ __align__(16) __bf16 Aks[AL_][128];
  __shared__ __align__(16) __bf16 Avs[AL_][128];

  const int qbase = blockIdx.x * 128;
  const int h = blockIdx.y;
  const int b = blockIdx.z;
  const int tid = threadIdx.x, lane = tid & 31, w = tid >> 5;
  const int lr = lane & 15, half = (lane >> 4) & 1;
  const int vs = *vsp;
  const float g2 = gate2[h];
  const float g1t = tanhf(gate1[h]);
  const float scale = 0.08838834764831845f;  // 1/sqrt(128)

  {  // stage Q tile (128 x 128 bf16)
    const int r = tid >> 1, seg = tid & 1;
    const uint4* src =
        (const uint4*)(qb + ((size_t)(b * S_ + qbase + r)) * D_ + h * HD_ + seg * 64);
    uint4* dst = (uint4*)(&Qs[r][seg * 64]);
#pragma unroll
    for (int j = 0; j < 8; j++) dst[j] = src[j];
  }
  for (int idx = tid; idx < AL_ * HD_; idx += 256) {  // stage adapter K/V rows
    const int al = idx >> 7, c = idx & 127;
    Aks[al][c] = kb[((size_t)b * SK_ + al) * D_ + h * HD_ + c];
    Avs[al][c] = vb[((size_t)b * SK_ + al) * D_ + h * HD_ + c];
  }
  __syncthreads();

  v16bf qf[4];
#pragma unroll
  for (int ks = 0; ks < 4; ks++) qf[ks] = ld_frag_a(&Qs[w * 16][ks * 32], 128, lane);

  v8f of[8];
#pragma unroll
  for (int t = 0; t < 8; t++) of[t] = (v8f){};
  float mrow[8], lrow[8];
#pragma unroll
  for (int r = 0; r < 8; r++) { mrow[r] = -3.0e38f; lrow[r] = 0.f; }

  const int nk = (qbase + 128) / 32;  // causal: only key tiles <= q tile
  for (int kt = 0; kt < nk; ++kt) {
    __syncthreads();  // previous iteration done with Ks/Vt
    {                 // stage K tile [32][128] and V tile transposed [128][40]
      const int r = tid >> 3, seg = tid & 7;
      const size_t grow =
          ((size_t)b * SK_ + AL_ + kt * 32 + r) * D_ + h * HD_ + seg * 16;
      const uint4* gk = (const uint4*)(kb + grow);
      uint4* lk = (uint4*)(&Ks[r][seg * 16]);
      lk[0] = gk[0]; lk[1] = gk[1];
      __bf16 tmp[16];
      ((uint4*)tmp)[0] = ((const uint4*)(vb + grow))[0];
      ((uint4*)tmp)[1] = ((const uint4*)(vb + grow))[1];
#pragma unroll
      for (int c = 0; c < 16; c++) Vt[seg * 16 + c][r] = tmp[c];
    }
    __syncthreads();

    // S = Q K^T  (M=16 rows of this wave, N=32 keys, K=128 hd)
    v8f sf[2];
#pragma unroll
    for (int nt = 0; nt < 2; nt++) {
      sf[nt] = (v8f){};
#pragma unroll
      for (int ks = 0; ks < 4; ks++)
        sf[nt] = wmma_bf16(qf[ks], ld_frag_b(&Ks[nt * 16][ks * 32], 128, lane),
                           sf[nt]);
    }
    // scale + causal mask + gate2 window (C layout: N = nt*16+lr, M = r+8*half)
#pragma unroll
    for (int nt = 0; nt < 2; nt++) {
      const int kpos = kt * 32 + nt * 16 + lr;
#pragma unroll
      for (int r = 0; r < 8; r++) {
        const int qpos = qbase + w * 16 + r + 8 * half;
        float s = sf[nt][r] * scale;
        if (kpos > qpos)
          s = -1.0e9f;
        else if (qpos >= vs + AL_ && kpos >= vs && kpos < vs + AL_)
          s += g2;
        sf[nt][r] = s;
      }
    }
    // online softmax per row, rescale O, emit P (bf16) into LDS
#pragma unroll
    for (int r = 0; r < 8; r++) {
      float mx = rmax16(fmaxf(sf[0][r], sf[1][r]));
      const float mnew = fmaxf(mrow[r], mx);
      const float alpha = __expf(mrow[r] - mnew);
      const float p0 = __expf(sf[0][r] - mnew);
      const float p1 = __expf(sf[1][r] - mnew);
      lrow[r] = lrow[r] * alpha + rsum16(p0 + p1);
      mrow[r] = mnew;
#pragma unroll
      for (int t = 0; t < 8; t++) of[t][r] *= alpha;
      Ps[w * 16 + r + 8 * half][lr] = (__bf16)p0;
      Ps[w * 16 + r + 8 * half][16 + lr] = (__bf16)p1;
    }
    // O += P V   (M=16, N=128 hd, K=32 keys); same-wave LDS RAW is in-order
    const v16bf pf = ld_frag_a(&Ps[w * 16][0], 32, lane);
#pragma unroll
    for (int t = 0; t < 8; t++)
      of[t] = wmma_bf16(pf, ld_frag_b(&Vt[t * 16][0], 40, lane), of[t]);
  }

  // normalize main softmax
#pragma unroll
  for (int r = 0; r < 8; r++) {
    const float inv = 1.0f / lrow[r];
#pragma unroll
    for (int t = 0; t < 8; t++) of[t][r] *= inv;
  }

  // adapter branch: separate softmax over AL columns, scaled by tanh(gate1)
#pragma unroll
  for (int r = 0; r < 8; r++) {
    const int row = w * 16 + r + 8 * half;
    const int hd0 = lr * 8;  // 16 lanes x 8 elems = 128-dim dot
    float sc[AL_];
    float mx = -1.0e30f;
#pragma unroll
    for (int al = 0; al < AL_; ++al) {
      float p = 0.f;
#pragma unroll
      for (int j = 0; j < 8; j++)
        p += (float)Qs[row][hd0 + j] * (float)Aks[al][hd0 + j];
      p = rsum16(p) * scale;
      sc[al] = p;
      mx = fmaxf(mx, p);
    }
    float den = 0.f;
#pragma unroll
    for (int al = 0; al < AL_; ++al) { sc[al] = __expf(sc[al] - mx); den += sc[al]; }
    const float g = g1t / den;
#pragma unroll
    for (int t = 0; t < 8; t++) {
      const int hdc = t * 16 + lr;
      float a = 0.f;
#pragma unroll
      for (int al = 0; al < AL_; ++al) a += sc[al] * (float)Avs[al][hdc];
      of[t][r] += g * a;
    }
  }

  // write attention output (bf16, [B][S][D] with col = h*HD + hd)
#pragma unroll
  for (int t = 0; t < 8; t++)
#pragma unroll
    for (int r = 0; r < 8; r++) {
      const int row = qbase + w * 16 + r + 8 * half;
      ob[((size_t)b * S_ + row) * D_ + h * HD_ + t * 16 + lr] = (__bf16)of[t][r];
    }
}

// ---------------------------------------------------------------------------
// launch
// ---------------------------------------------------------------------------
extern "C" void kernel_launch(void* const* d_in, const int* in_sizes, int n_in,
                              void* d_out, int out_size, void* d_ws,
                              size_t ws_size, hipStream_t stream) {
  const float* x       = (const float*)d_in[0];
  const float* adapter = (const float*)d_in[1];
  const float* cosb    = (const float*)d_in[2];
  const float* sinb    = (const float*)d_in[3];
  // d_in[4] = mask: recomputed analytically (causal), unused
  const float* wq      = (const float*)d_in[5];
  const float* wk      = (const float*)d_in[6];
  const float* wv      = (const float*)d_in[7];
  const float* wo      = (const float*)d_in[8];
  const float* gate1   = (const float*)d_in[9];
  const float* gate2   = (const float*)d_in[10];
  const int*   vsp     = (const int*)d_in[11];

  char* ws = (char*)d_ws;
  size_t off = 0;
  auto alloc = [&](size_t bytes) -> void* {
    void* p = ws + off;
    off += (bytes + 255) & ~(size_t)255;
    return p;
  };
  __bf16* xb  = (__bf16*)alloc((size_t)B_ * S_ * D_ * 2);
  __bf16* wqT = (__bf16*)alloc((size_t)D_ * D_ * 2);
  __bf16* wkT = (__bf16*)alloc((size_t)D_ * D_ * 2);
  __bf16* wvT = (__bf16*)alloc((size_t)D_ * D_ * 2);
  __bf16* woT = (__bf16*)alloc((size_t)D_ * D_ * 2);
  __bf16* qb  = (__bf16*)alloc((size_t)B_ * S_ * D_ * 2);
  __bf16* kb  = (__bf16*)alloc((size_t)B_ * SK_ * D_ * 2);
  __bf16* vb  = (__bf16*)alloc((size_t)B_ * SK_ * D_ * 2);
  __bf16* ab  = (__bf16*)alloc((size_t)B_ * S_ * D_ * 2);

  const int nX = B_ * S_ * D_;
  convert_f32_bf16<<<nX / 256, 256, 0, stream>>>(x, xb, nX);
  dim3 tg(D_ / 32, D_ / 32);
  transpose_f32_bf16<<<tg, 256, 0, stream>>>(wq, wqT, D_, D_);
  transpose_f32_bf16<<<tg, 256, 0, stream>>>(wk, wkT, D_, D_);
  transpose_f32_bf16<<<tg, 256, 0, stream>>>(wv, wvT, D_, D_);
  transpose_f32_bf16<<<tg, 256, 0, stream>>>(wo, woT, D_, D_);

  const dim3 gg(D_ / 128, (B_ * S_) / 128);
  gemm_wmma<false><<<gg, 256, 0, stream>>>(xb, wqT, qb, B_ * S_, D_, D_, S_, S_, 0);
  gemm_wmma<false><<<gg, 256, 0, stream>>>(xb, wkT, kb, B_ * S_, D_, D_, S_, SK_, AL_);
  gemm_wmma<false><<<gg, 256, 0, stream>>>(xb, wvT, vb, B_ * S_, D_, D_, S_, SK_, AL_);

  adapter_kv<<<(AL_ * D_ + 255) / 256, 256, 0, stream>>>(adapter, wkT, wvT, kb, vb);

  const int nPairs = B_ * S_ * (D_ / 2);
  rope_kernel<<<nPairs / 256, 256, 0, stream>>>(qb, cosb, sinb, S_, 0);
  rope_kernel<<<nPairs / 256, 256, 0, stream>>>(kb, cosb, sinb, SK_, AL_);

  attn_kernel<<<dim3(S_ / 128, H_, B_), 256, 0, stream>>>(qb, kb, vb, ab, gate1,
                                                          gate2, vsp);

  gemm_wmma<true><<<gg, 256, 0, stream>>>(ab, woT, d_out, B_ * S_, D_, D_, S_, S_, 0);
}
